// EdgeMessagePassingLayer_45655502356584
// MI455X (gfx1250) — 8-bit matrix operand order repaired and checked by reading, compiled, not run on hardware
//
#include <hip/hip_runtime.h>
#include <hip/hip_bf16.h>
#include <stdint.h>

#define NN 4096

typedef __attribute__((ext_vector_type(8))) int v8i;
typedef int i4 __attribute__((vector_size(16)));

#if __has_builtin(__builtin_amdgcn_global_load_async_to_lds_b128) && \
    __has_builtin(__builtin_amdgcn_s_wait_asynccnt)
#define HAVE_ASYNC_LDS 1
#else
#define HAVE_ASYNC_LDS 0
#endif

union Frag {
  v8i       v;
  int       i[8];
  uint32_t  u[8];
  uint2     u2[4];
  uint4     u4[2];
};

__global__ __launch_bounds__(256) void pack_kernel(const float* __restrict__ A,
                                                   uint8_t* __restrict__ M) {
  size_t base = ((size_t)blockIdx.x * 256 + threadIdx.x) * 4;
  uint32_t packed = 0;
#pragma unroll
  for (int t = 0; t < 4; ++t)
    packed |= (A[base + t] != 0.0f ? 1u : 0u) << (8 * t);
  *(uint32_t*)(M + base) = packed;
}

__global__ __launch_bounds__(256) void rowsum_kernel(const float* __restrict__ E,
                                                     float* __restrict__ e) {
  __shared__ float red[256];
  const float* p = E + (size_t)blockIdx.x * NN;
  float s = 0.f;
  for (int c = threadIdx.x; c < NN; c += 256) s += p[c];
  red[threadIdx.x] = s;
  __syncthreads();
  for (int off = 128; off > 0; off >>= 1) {
    if ((int)threadIdx.x < off) red[threadIdx.x] += red[threadIdx.x + off];
    __syncthreads();
  }
  if (threadIdx.x == 0) e[blockIdx.x] = red[0];
}

__device__ __forceinline__ void load_a(const uint8_t* ldsA, int nl, int h, int k0,
                                       Frag& a) {
  const uint8_t* arow = ldsA + (size_t)nl * NN + k0 + h * 16;
  a.u4[0] = *(const uint4*)(arow);
  a.u4[1] = *(const uint4*)(arow + 32);
}

__device__ __forceinline__ void load_b(const uint8_t* brow, int k0, Frag& b) {
  b.u4[0] = *(const uint4*)(brow + k0);
  b.u4[1] = *(const uint4*)(brow + k0 + 32);
}

__global__ __launch_bounds__(512) void linegraph_kernel(const uint8_t* __restrict__ M,
                                                        const float* __restrict__ e,
                                                        float* __restrict__ agg) {
  __shared__ __align__(16) uint8_t ldsA[16 * NN];
  const int iBase = blockIdx.x * 16;
  const int tid   = threadIdx.x;
  const int wave  = tid >> 5;
  const int lane  = tid & 31;
  const int nl    = lane & 15;
  const int h     = lane >> 4;

#if HAVE_ASYNC_LDS
  {
    uint8_t* src = const_cast<uint8_t*>(M) + (size_t)iBase * NN;
    for (int t = tid; t < 16 * NN / 16; t += 512) {
      __builtin_amdgcn_global_load_async_to_lds_b128(
          (__attribute__((address_space(1))) i4*)(src + t * 16),
          (__attribute__((address_space(3))) i4*)(ldsA + t * 16),
          0, 0);
    }
    __builtin_amdgcn_s_wait_asynccnt(0);
  }
#else
  {
    const uint4* src = (const uint4*)(M + (size_t)iBase * NN);
    uint4*       dst = (uint4*)ldsA;
    for (int t = tid; t < 16 * NN / 16; t += 512) dst[t] = src[t];
  }
#endif
  __syncthreads();

  int di[8];
#pragma unroll
  for (int r = 0; r < 8; ++r)
    di[r] = ldsA[(size_t)(h * 8 + r) * NN + iBase + h * 8 + r];

  float acc[8] = {0.f, 0.f, 0.f, 0.f, 0.f, 0.f, 0.f, 0.f};

#pragma clang loop unroll(disable)
  for (int jj = 0; jj < 8; ++jj) {
    const int jBase = (jj * 32 + wave * 2) * 16;
    const int j0    = jBase + nl;
    const int j1    = j0 + 16;
    const uint8_t* brow0 = M + (size_t)j0 * NN + (h << 4);
    const uint8_t* brow1 = M + (size_t)j1 * NN + (h << 4);

    Frag c0, c1;
    c0.v = (v8i){0, 0, 0, 0, 0, 0, 0, 0};
    c1.v = (v8i){0, 0, 0, 0, 0, 0, 0, 0};

    Frag aP, b0P, b1P, aQ, b0Q, b1Q;
    load_a(ldsA, nl, h, 0, aP);
    load_b(brow0, 0, b0P);
    load_b(brow1, 0, b1P);
#pragma clang loop unroll(disable)
    for (int k0 = 0; k0 < NN - 128; k0 += 128) {
      load_a(ldsA, nl, h, k0 + 64, aQ);
      load_b(brow0, k0 + 64, b0Q);
      load_b(brow1, k0 + 64, b1Q);
      c0.v = __builtin_amdgcn_wmma_i32_16x16x64_iu8(false, aP.v, false, b0P.v,
                                                    c0.v, false, false);
      c1.v = __builtin_amdgcn_wmma_i32_16x16x64_iu8(false, aP.v, false, b1P.v,
                                                    c1.v, false, false);
      load_a(ldsA, nl, h, k0 + 128, aP);
      load_b(brow0, k0 + 128, b0P);
      load_b(brow1, k0 + 128, b1P);
      c0.v = __builtin_amdgcn_wmma_i32_16x16x64_iu8(false, aQ.v, false, b0Q.v,
                                                    c0.v, false, false);
      c1.v = __builtin_amdgcn_wmma_i32_16x16x64_iu8(false, aQ.v, false, b1Q.v,
                                                    c1.v, false, false);
    }
    {
      load_a(ldsA, nl, h, NN - 64, aQ);
      load_b(brow0, NN - 64, b0Q);
      load_b(brow1, NN - 64, b1Q);
      c0.v = __builtin_amdgcn_wmma_i32_16x16x64_iu8(false, aP.v, false, b0P.v,
                                                    c0.v, false, false);
      c1.v = __builtin_amdgcn_wmma_i32_16x16x64_iu8(false, aP.v, false, b1P.v,
                                                    c1.v, false, false);
      c0.v = __builtin_amdgcn_wmma_i32_16x16x64_iu8(false, aQ.v, false, b0Q.v,
                                                    c0.v, false, false);
      c1.v = __builtin_amdgcn_wmma_i32_16x16x64_iu8(false, aQ.v, false, b1Q.v,
                                                    c1.v, false, false);
    }

    {
      const int      dj  = M[(size_t)j0 * NN + j0];
      const uint64_t mji = *(const uint64_t*)(M + (size_t)j0 * NN + iBase + h * 8);
      const float    ej  = e[j0];
#pragma unroll
      for (int r = 0; r < 8; ++r) {
        const int i    = iBase + h * 8 + r;
        const int mij  = ldsA[(size_t)(h * 8 + r) * NN + j0];
        const int mj_i = (int)((mji >> (8 * r)) & 0xff);
        const int cnt  = c0.i[r] - di[r] * mj_i - mij * dj;
        if (cnt > 0 && i != j0) acc[r] += ej;
      }
    }
    {
      const int      dj  = M[(size_t)j1 * NN + j1];
      const uint64_t mji = *(const uint64_t*)(M + (size_t)j1 * NN + iBase + h * 8);
      const float    ej  = e[j1];
#pragma unroll
      for (int r = 0; r < 8; ++r) {
        const int i    = iBase + h * 8 + r;
        const int mij  = ldsA[(size_t)(h * 8 + r) * NN + j1];
        const int mj_i = (int)((mji >> (8 * r)) & 0xff);
        const int cnt  = c1.i[r] - di[r] * mj_i - mij * dj;
        if (cnt > 0 && i != j1) acc[r] += ej;
      }
    }
  }

#pragma unroll
  for (int off = 1; off <= 8; off <<= 1)
#pragma unroll
    for (int r = 0; r < 8; ++r) acc[r] += __shfl_xor(acc[r], off, 32);

  __syncthreads();
  float* fred = (float*)ldsA;
  if (nl == 0) {
#pragma unroll
    for (int r = 0; r < 8; ++r) fred[wave * 16 + h * 8 + r] = acc[r];
  }
  __syncthreads();
  if (tid < 16) {
    float s = 0.f;
    for (int w = 0; w < 16; ++w) s += fred[w * 16 + tid];
    agg[iBase + tid] = s;
  }
}

__global__ __launch_bounds__(256) void outproj_kernel(const float* __restrict__ W,
                                                      const float* __restrict__ bias,
                                                      const float* __restrict__ agg,
                                                      float* __restrict__ out) {
  __shared__ float red[256];
  const float* wr = W + (size_t)blockIdx.x * NN;
  float s = 0.f;
  for (int c = threadIdx.x; c < NN; c += 256) s += wr[c] * agg[c];
  red[threadIdx.x] = s;
  __syncthreads();
  for (int off = 128; off > 0; off >>= 1) {
    if ((int)threadIdx.x < off) red[threadIdx.x] += red[threadIdx.x + off];
    __syncthreads();
  }
  if (threadIdx.x == 0) out[blockIdx.x] = red[0] + bias[blockIdx.x];
}

extern "C" void kernel_launch(void* const* d_in, const int* in_sizes, int n_in,
                              void* d_out, int out_size, void* d_ws, size_t ws_size,
                              hipStream_t stream) {
  (void)in_sizes; (void)n_in; (void)out_size; (void)ws_size;
  const float* A = (const float*)d_in[0];
  const float* E = (const float*)d_in[1];
  const float* W = (const float*)d_in[2];
  const float* b = (const float*)d_in[3];
  float* out = (float*)d_out;

  uint8_t* Mu8 = (uint8_t*)d_ws;
  float*   e   = (float*)((uint8_t*)d_ws + (size_t)NN * NN);
  float*   agg = e + NN;

  pack_kernel<<<(NN * NN / 4) / 256, 256, 0, stream>>>(A, Mu8);
  rowsum_kernel<<<NN, 256, 0, stream>>>(E, e);
  linegraph_kernel<<<NN / 16, 512, 0, stream>>>(Mu8, e, agg);
  outproj_kernel<<<NN, 256, 0, stream>>>(W, b, agg, out);
}
